// DecoderScannedRNN_429496729929
// MI455X (gfx1250) — compile-verified, matrix operand order checked by reading, not run on hardware
//
#include <hip/hip_runtime.h>
#include <hip/hip_bf16.h>
#include <math.h>

// Problem constants (match reference)
#define T_STEPS 256
#define BATCH   64
#define DIM     1024
#define G3      (3 * DIM)
#define NWG_SCAN 64
#define COLS_PER_WG (DIM / NWG_SCAN)   // 16 hidden columns per workgroup
#define WROWS   (3 * COLS_PER_WG)      // 48 W^T rows resident per WG

// ---- workspace layout (bytes, all 256B aligned) ----
#define WIT_OFF   0                                   // 3072*1024*2 = 6 MB
#define WHT_OFF   (6291456)                           // 6 MB
#define INSB_OFF  (12582912)                          // T*B*D*2 = 32 MB
#define HBF_OFF   (46137344)                          // 2 * B*D * 2 = 256 KB
#define CNT_OFF   (46399488)                          // barrier counter
#define XP_OFF    (46399744)                          // T*B*3D*4 = 192 MB
// total ~237 MB

typedef __attribute__((ext_vector_type(16))) __bf16 v16bf;
typedef __attribute__((ext_vector_type(8)))  float  v8f;

union V16 { v16bf v; uint4 q[2]; };

__device__ __forceinline__ unsigned short f32_to_bf16(float f) {
  unsigned u = __float_as_uint(f);
  unsigned rnd = 0x7FFFu + ((u >> 16) & 1u);   // round-to-nearest-even
  return (unsigned short)((u + rnd) >> 16);
}

__device__ __forceinline__ float sigmoidf_(float x) {
  return 1.0f / (1.0f + __expf(-x));
}

__device__ __forceinline__ v8f wmma_bf16(const V16& a, const V16& b, v8f c) {
  // D = A(16x32) * B(32x16) + C, f32 accumulate
  return __builtin_amdgcn_wmma_f32_16x16x32_bf16(
      /*neg_a=*/false, a.v, /*neg_b=*/false, b.v,
      /*c_mod=*/(short)0, c, /*reuse_a=*/false, /*reuse_b=*/false);
}

// Device-wide barrier (persistent kernel; 64 co-resident workgroups)
__device__ __forceinline__ void grid_barrier(unsigned* cnt, unsigned target) {
  __threadfence();
  __syncthreads();
  if (threadIdx.x == 0) {
    atomicAdd(cnt, 1u);
    while (__hip_atomic_load(cnt, __ATOMIC_ACQUIRE, __HIP_MEMORY_SCOPE_AGENT) < target)
      __builtin_amdgcn_s_sleep(2);
  }
  __syncthreads();
}

// ---------------------------------------------------------------------------
// Small prep kernels
// ---------------------------------------------------------------------------
__global__ void k_zero_cnt(unsigned* cnt) {
  if (threadIdx.x == 0) *cnt = 0u;
}

// f32 -> bf16 flat convert (grid-stride)
__global__ void k_cvt(const float* __restrict__ src,
                      unsigned short* __restrict__ dst, int n) {
  int i = blockIdx.x * blockDim.x + threadIdx.x;
  int stride = gridDim.x * blockDim.x;
  for (; i < n; i += stride) dst[i] = f32_to_bf16(src[i]);
}

// W [1024][3072] f32  ->  W^T [3072][1024] bf16  (writes coalesced)
__global__ void k_transp(const float* __restrict__ W,
                         unsigned short* __restrict__ WT) {
  int i = blockIdx.x * blockDim.x + threadIdx.x;
  if (i < G3 * DIM) {
    int d = i & (DIM - 1);
    int c = i >> 10;
    WT[i] = f32_to_bf16(W[d * G3 + c]);
  }
}

// ---------------------------------------------------------------------------
// Phase 1: x_proj = ins @ W_i  (bias added later in the scan epilogue)
//   A = ins_bf16 [16384 x 1024] row-major, Bt = WiT [3072 x 1024] row-major
//   Each block (256 thr = 8 waves) computes a 64x64 f32 output tile;
//   each wave computes two 16x16 tiles sharing its A loads.
// ---------------------------------------------------------------------------
__global__ __launch_bounds__(256, 2)
void k_xproj(const unsigned short* __restrict__ insb,
             const unsigned short* __restrict__ WiT,
             float* __restrict__ xp) {
  const int w    = threadIdx.x >> 5;
  const int lane = threadIdx.x & 31;
  const int mtile = w & 3;         // 4 m-tiles of 16 rows
  const int npair = w >> 2;        // 0..1, each owns 2 n-tiles
  const int row0 = blockIdx.x * 64 + mtile * 16;
  const int col0 = blockIdx.y * 64 + npair * 32;
  const int half = lane >> 4;
  const int r    = lane & 15;

  const unsigned short* a  = insb + (size_t)(row0 + r) * DIM + half * 8;
  const unsigned short* b0 = WiT  + (size_t)(col0 + r) * DIM + half * 8;
  const unsigned short* b1 = WiT  + (size_t)(col0 + 16 + r) * DIM + half * 8;

  v8f acc0 = {}, acc1 = {};
#pragma unroll 4
  for (int k = 0; k < DIM; k += 32) {
    V16 av, bv0, bv1;
    av.q[0]  = *(const uint4*)(a + k);
    av.q[1]  = *(const uint4*)(a + k + 16);
    bv0.q[0] = *(const uint4*)(b0 + k);
    bv0.q[1] = *(const uint4*)(b0 + k + 16);
    bv1.q[0] = *(const uint4*)(b1 + k);
    bv1.q[1] = *(const uint4*)(b1 + k + 16);
    acc0 = wmma_bf16(av, bv0, acc0);
    acc1 = wmma_bf16(av, bv1, acc1);
  }
  // C/D layout: acc[i] holds element (m = i + 8*half, n = lane&15)
#pragma unroll
  for (int i = 0; i < 8; ++i) {
    int rr = i + 8 * half;
    xp[(size_t)(row0 + rr) * G3 + col0 + r]      = acc0[i];
    xp[(size_t)(row0 + rr) * G3 + col0 + 16 + r] = acc1[i];
  }
}

// ---------------------------------------------------------------------------
// Phase 2: persistent GRU scan.
//   64 workgroups x 384 threads (12 waves). WG owns 16 hidden cols (all 3
//   gates). Its 48-row W_h^T slice (96 KB bf16) is parked in LDS once and
//   read via ds_load_b128 every step, while the ping-pong bf16 h state
//   streams through the VMEM path — splitting operand traffic across the
//   two independent datapaths. Wave (mtile = w&3, gate g = w>>2) computes
//   one 16x16 tile of h @ W_h-slice per step with two independent WMMA
//   accumulation chains over K. Gate mix via LDS; one device barrier/step.
// ---------------------------------------------------------------------------
__global__ __launch_bounds__(384, 1)
void k_gru_scan(const unsigned short* __restrict__ WhT,
                const float* __restrict__ xp,
                const unsigned char* __restrict__ dones,
                const float* __restrict__ hiddens,
                const float* __restrict__ init_carry,
                const float* __restrict__ b_i,
                const float* __restrict__ b_hn,
                unsigned short* __restrict__ hbf,   // 2 x [B*D] bf16
                float* __restrict__ out,            // [B*D final | T*B*D ys]
                unsigned* __restrict__ cnt) {
  __shared__ __align__(16) unsigned short sh_w[WROWS][DIM]; // 96 KB weights
  __shared__ float sh_pre[4][2][16][16];           // [mtile][0:z,1:hWn][rr][cc]
  __shared__ float sh_heff[BATCH][COLS_PER_WG];    // post-done h for this step

  const int tid   = threadIdx.x;
  const int lane  = tid & 31;
  const int w     = tid >> 5;       // 0..11
  const int mtile = w & 3;
  const int g     = w >> 2;         // 0=r, 1=z, 2=n
  const int colbase = blockIdx.x * COLS_PER_WG;
  const int half  = lane >> 4;
  const int r     = lane & 15;

  float* ys = out + BATCH * DIM;

  // ---- park this WG's W_h^T slice in LDS (once, uint4 copies)
  for (int idx = tid; idx < WROWS * (DIM / 8); idx += blockDim.x) {
    int j  = idx / (DIM / 8);            // 0..47: (gate<<4) | cc
    int kk = (idx % (DIM / 8)) * 8;
    int grow = (j >> 4) * DIM + colbase + (j & 15);
    *(uint4*)&sh_w[j][kk] = *(const uint4*)(WhT + (size_t)grow * DIM + kk);
  }

  // ---- h_eff entering step 0 = where(done[0], hiddens[0], init_carry)
  for (int idx = tid; idx < BATCH * COLS_PER_WG; idx += blockDim.x) {
    int m  = idx >> 4;
    int cc = idx & (COLS_PER_WG - 1);
    int col = colbase + cc;
    float h0 = init_carry[m * DIM + col];
    float he = dones[m] ? hiddens[(size_t)m * DIM + col] : h0;
    sh_heff[m][cc] = he;
    hbf[m * DIM + col] = f32_to_bf16(he);   // buffer 0
  }
  unsigned sync_no = 1;
  grid_barrier(cnt, sync_no * NWG_SCAN);   // also covers the LDS preload

  // This wave's weight tile base in LDS: rows [g*16, +16), contiguous in K
  const unsigned short* Bt = &sh_w[g * 16 + r][half * 8];

  for (int t = 0; t < T_STEPS; ++t) {
    // ---- 16x16 tile of h @ W_h-slice, K = 1024, two independent chains
    const unsigned short* A =
        hbf + (t & 1) * (BATCH * DIM) + (size_t)(mtile * 16 + r) * DIM + half * 8;
    v8f acc0 = {}, acc1 = {};
#pragma unroll 4
    for (int k = 0; k < DIM / 2; k += 32) {
      V16 av0, bv0, av1, bv1;
      av0.q[0] = *(const uint4*)(A + k);
      av0.q[1] = *(const uint4*)(A + k + 16);
      bv0.q[0] = *(const uint4*)(Bt + k);
      bv0.q[1] = *(const uint4*)(Bt + k + 16);
      av1.q[0] = *(const uint4*)(A + (DIM / 2) + k);
      av1.q[1] = *(const uint4*)(A + (DIM / 2) + k + 16);
      bv1.q[0] = *(const uint4*)(Bt + (DIM / 2) + k);
      bv1.q[1] = *(const uint4*)(Bt + (DIM / 2) + k + 16);
      acc0 = wmma_bf16(av0, bv0, acc0);
      acc1 = wmma_bf16(av1, bv1, acc1);
    }
    v8f acc = acc0 + acc1;

    if (g > 0) {   // export z-pre and hWn tiles for the combiner waves
#pragma unroll
      for (int i = 0; i < 8; ++i)
        sh_pre[mtile][g - 1][i + 8 * half][r] = acc[i];
    }
    __syncthreads();

    if (g == 0) {  // combiner waves (one per mtile) hold the r-gate tile
#pragma unroll
      for (int i = 0; i < 8; ++i) {
        int rr = i + 8 * half;
        int cc = r;
        int m  = mtile * 16 + rr;
        int col = colbase + cc;
        const float* xrow = xp + (size_t)(t * BATCH + m) * G3;
        float rg = sigmoidf_(xrow[col] + b_i[col] + acc[i]);
        float zg = sigmoidf_(xrow[DIM + col] + b_i[DIM + col] +
                             sh_pre[mtile][0][rr][cc]);
        float ng = tanhf(xrow[2 * DIM + col] + b_i[2 * DIM + col] +
                         rg * (sh_pre[mtile][1][rr][cc] + b_hn[col]));
        float hprev = sh_heff[m][cc];
        float hnew  = (1.0f - zg) * ng + zg * hprev;
        ys[(size_t)(t * BATCH + m) * DIM + col] = hnew;
        if (t == T_STEPS - 1) {
          out[m * DIM + col] = hnew;            // final carry
        } else {
          // fuse next step's done-reset into this writeback
          float he = dones[(t + 1) * BATCH + m]
                         ? hiddens[(size_t)((t + 1) * BATCH + m) * DIM + col]
                         : hnew;
          sh_heff[m][cc] = he;
          hbf[((t + 1) & 1) * (BATCH * DIM) + m * DIM + col] = f32_to_bf16(he);
        }
      }
    }

    if (t + 1 < T_STEPS) {
      ++sync_no;
      grid_barrier(cnt, sync_no * NWG_SCAN);
    }
  }
}

// ---------------------------------------------------------------------------
extern "C" void kernel_launch(void* const* d_in, const int* in_sizes, int n_in,
                              void* d_out, int out_size, void* d_ws, size_t ws_size,
                              hipStream_t stream) {
  (void)in_sizes; (void)n_in; (void)out_size; (void)ws_size;

  const float*         ins        = (const float*)d_in[0];   // [T,B,D]
  const float*         hiddens    = (const float*)d_in[1];   // [T,B,D]
  const unsigned char* dones      = (const unsigned char*)d_in[2]; // [T,B] bool
  const float*         init_carry = (const float*)d_in[3];   // [B,D]
  const float*         W_i        = (const float*)d_in[4];   // [D,3D]
  const float*         W_h        = (const float*)d_in[5];   // [D,3D]
  const float*         b_i        = (const float*)d_in[6];   // [3D]
  const float*         b_hn       = (const float*)d_in[7];   // [D]

  char* ws = (char*)d_ws;
  unsigned short* WiT  = (unsigned short*)(ws + WIT_OFF);
  unsigned short* WhT  = (unsigned short*)(ws + WHT_OFF);
  unsigned short* insb = (unsigned short*)(ws + INSB_OFF);
  unsigned short* hbf  = (unsigned short*)(ws + HBF_OFF);
  unsigned*       cnt  = (unsigned*)(ws + CNT_OFF);
  float*          xp   = (float*)(ws + XP_OFF);
  float*          out  = (float*)d_out;

  // 0) reset device barrier counter (fresh every launch/replay)
  k_zero_cnt<<<dim3(1), dim3(32), 0, stream>>>(cnt);

  // 1) ins f32 -> bf16
  k_cvt<<<dim3(4096), dim3(256), 0, stream>>>(ins, insb, T_STEPS * BATCH * DIM);

  // 2) transpose + convert weights: W^T [3D,1024] bf16 (contiguous-K B tiles)
  k_transp<<<dim3((G3 * DIM) / 256), dim3(256), 0, stream>>>(W_i, WiT);
  k_transp<<<dim3((G3 * DIM) / 256), dim3(256), 0, stream>>>(W_h, WhT);

  // 3) hoisted input projection: x_proj = ins @ W_i   (103 GFLOP, parallel)
  k_xproj<<<dim3((T_STEPS * BATCH) / 64, G3 / 64), dim3(256), 0, stream>>>(
      insb, WiT, xp);

  // 4) persistent sequential scan (103 GFLOP over 256 dependent steps)
  k_gru_scan<<<dim3(NWG_SCAN), dim3(384), 0, stream>>>(
      WhT, xp, dones, hiddens, init_carry, b_i, b_hn, hbf, out, cnt);
}
